// QuantizedMatMul_56788057588243
// MI455X (gfx1250) — 8-bit matrix operand order repaired and checked by reading, compiled, not run on hardware
//
#include <hip/hip_runtime.h>
#include <stdint.h>

typedef __attribute__((ext_vector_type(8))) int   v8i;
typedef __attribute__((ext_vector_type(8))) float v8f;
typedef __attribute__((ext_vector_type(4))) float v4f;

#define S_DIM 2048
#define D_DIM 64
#define BH    64
#define PER_TENSOR ((size_t)BH * S_DIM * D_DIM)

__device__ __forceinline__ unsigned f32_to_e4m3(float x) {
    unsigned u    = __float_as_uint(x);
    unsigned sign = (u >> 24) & 0x80u;
    float ax      = fabsf(x);
    if (!(ax < 448.0f)) {
        return sign | 0x7Eu;
    }
    int ee = (int)((u >> 23) & 0xFFu) - 127;
    int e  = ee < -6 ? -6 : ee;
    float scale = exp2f((float)(3 - e));
    int q = (int)rintf(ax * scale);
    if (q >= 16) { e += 1; q = 8; }
    unsigned expf_, mant;
    if (q < 8) { expf_ = 0u; mant = (unsigned)q; }
    else       { expf_ = (unsigned)(e + 7); mant = (unsigned)(q - 8); }
    return sign | (expf_ << 3) | mant;
}

__global__ __launch_bounds__(256) void amax_kernel(const float* __restrict__ in,
                                                   long n4, unsigned* __restrict__ out) {
    const v4f* in4 = (const v4f*)in;
    unsigned m = 0u;
    long i = (long)blockIdx.x * blockDim.x + threadIdx.x;
    long stride = (long)gridDim.x * blockDim.x;
    for (; i < n4; i += stride) {
        v4f v = in4[i];
        unsigned b0 = __float_as_uint(fabsf(v.x));
        unsigned b1 = __float_as_uint(fabsf(v.y));
        unsigned b2 = __float_as_uint(fabsf(v.z));
        unsigned b3 = __float_as_uint(fabsf(v.w));
        m = max(m, max(max(b0, b1), max(b2, b3)));
    }
    __shared__ unsigned red[256];
    red[threadIdx.x] = m;
    __syncthreads();
    for (int s = 128; s > 0; s >>= 1) {
        if ((int)threadIdx.x < s) red[threadIdx.x] = max(red[threadIdx.x], red[threadIdx.x + s]);
        __syncthreads();
    }
    if (threadIdx.x == 0) atomicMax(out, red[0]);
}

__global__ __launch_bounds__(256) void quant_a_kernel(const float* __restrict__ in,
                                                      unsigned* __restrict__ out,
                                                      const unsigned* __restrict__ amax_bits,
                                                      long n4) {
    long i = (long)blockIdx.x * blockDim.x + threadIdx.x;
    if (i >= n4) return;
    float amax = __uint_as_float(*amax_bits);
    float inv  = amax > 0.0f ? 448.0f / amax : 1.0f;
    v4f v = __builtin_nontemporal_load((const v4f*)in + i);
    unsigned p =  f32_to_e4m3(v.x * inv)
               | (f32_to_e4m3(v.y * inv) << 8)
               | (f32_to_e4m3(v.z * inv) << 16)
               | (f32_to_e4m3(v.w * inv) << 24);
    out[i] = p;
}

__global__ __launch_bounds__(256) void quant_b_kernel(const float* __restrict__ in,
                                                      unsigned* __restrict__ out,
                                                      const unsigned* __restrict__ amax_bits,
                                                      long nthreads) {
    long g = (long)blockIdx.x * blockDim.x + threadIdx.x;
    if (g >= nthreads) return;
    float amax = __uint_as_float(*amax_bits);
    float inv  = amax > 0.0f ? 448.0f / amax : 1.0f;
    int bh = (int)(g >> 13);
    int r  = (int)(g & 8191);
    int dq = r >> 9;
    int sq = r & 511;
    const float* ib = in + ((long)bh << 17) + ((long)(dq << 2) * S_DIM) + ((long)sq << 2);
    unsigned w0 = 0, w1 = 0, w2 = 0, w3 = 0;
#pragma unroll
    for (int j = 0; j < 4; ++j) {
        v4f v = __builtin_nontemporal_load((const v4f*)(ib + (long)j * S_DIM));
        w0 |= f32_to_e4m3(v.x * inv) << (8 * j);
        w1 |= f32_to_e4m3(v.y * inv) << (8 * j);
        w2 |= f32_to_e4m3(v.z * inv) << (8 * j);
        w3 |= f32_to_e4m3(v.w * inv) << (8 * j);
    }
    unsigned* ob = out + ((long)bh << 15) + ((long)(sq << 2) << 4) + dq;
    ob[0 << 4] = w0;
    ob[1 << 4] = w1;
    ob[2 << 4] = w2;
    ob[3 << 4] = w3;
}

__global__ __launch_bounds__(256) void gemm_fp8_kernel(const unsigned char* __restrict__ Aq,
                                                       const unsigned char* __restrict__ Bq,
                                                       const unsigned* __restrict__ amax_a,
                                                       const unsigned* __restrict__ amax_b,
                                                       float* __restrict__ out) {
    const int lane = threadIdx.x & 31;
    const int wave = threadIdx.x >> 5;
    const int bh   = blockIdx.z;
    const int m0   = blockIdx.y << 4;
    const int n0   = (blockIdx.x * 8 + wave) << 6;
    const int half = lane >> 4;
    const int l15  = lane & 15;

    const unsigned char* Ab = Aq + ((size_t)bh << 17);
    const unsigned char* Bb = Bq + ((size_t)bh << 17);

    const unsigned char* aptr = Ab + ((size_t)(m0 + l15) << 6) + (half << 4);
    v8i a;
    {
        int4 a0 = *(const int4*)(aptr +  0);
        int4 a1 = *(const int4*)(aptr + 32);
        a[0] = a0.x; a[1] = a0.y; a[2] = a0.z; a[3] = a0.w;
        a[4] = a1.x; a[5] = a1.y; a[6] = a1.z; a[7] = a1.w;
    }

    const float scale = (__uint_as_float(*amax_a) * (1.0f / 448.0f)) *
                        (__uint_as_float(*amax_b) * (1.0f / 448.0f));

    float* obase = out + ((size_t)bh << 22);

#pragma unroll
    for (int t = 0; t < 4; ++t) {
        const int n = n0 + (t << 4) + l15;
        const unsigned char* bptr = Bb + ((size_t)n << 6) + (half << 4);
        v8i b;
        {
            int4 b0 = *(const int4*)(bptr);
            int4 b1 = *(const int4*)(bptr + 32);
            b[0] = b0.x; b[1] = b0.y; b[2] = b0.z; b[3] = b0.w;
            b[4] = b1.x; b[5] = b1.y; b[6] = b1.z; b[7] = b1.w;
        }
        v8f c = {};
        c = __builtin_amdgcn_wmma_f32_16x16x64_fp8_fp8(a, b, (short)0, c, false, false);
#pragma unroll
        for (int r = 0; r < 8; ++r) {
            const int row = m0 + r + (half << 3);
            __builtin_nontemporal_store(c[r] * scale, &obase[(size_t)row * S_DIM + n]);
        }
    }
}

extern "C" void kernel_launch(void* const* d_in, const int* in_sizes, int n_in,
                              void* d_out, int out_size, void* d_ws, size_t ws_size,
                              hipStream_t stream) {
    (void)in_sizes; (void)n_in; (void)out_size; (void)ws_size;
    const float* in1 = (const float*)d_in[0];
    const float* in2 = (const float*)d_in[1];
    float* out = (float*)d_out;

    unsigned char* ws = (unsigned char*)d_ws;
    unsigned* amaxA = (unsigned*)ws;
    unsigned* amaxB = (unsigned*)(ws + 4);
    unsigned char* Aq = ws + 256;
    unsigned char* Bq = ws + 256 + PER_TENSOR;

    hipMemsetAsync(ws, 0, 256, stream);

    const long n  = (long)PER_TENSOR;
    const long n4 = n / 4;

    amax_kernel<<<2048, 256, 0, stream>>>(in1, n4, amaxA);
    amax_kernel<<<2048, 256, 0, stream>>>(in2, n4, amaxB);

    quant_a_kernel<<<(unsigned)((n4 + 255) / 256), 256, 0, stream>>>(in1, (unsigned*)Aq, amaxA, n4);

    const long nbt = n / 16;
    quant_b_kernel<<<(unsigned)((nbt + 255) / 256), 256, 0, stream>>>(in2, (unsigned*)Bq, amaxB, nbt);

    dim3 grid(4, 128, 64);
    gemm_fp8_kernel<<<grid, 256, 0, stream>>>(Aq, Bq, amaxA, amaxB, out);
}
